// GroupedQueryAttention_75642964017415
// MI455X (gfx1250) — matrix operand order repaired and checked by reading, not compiled, not run on hardware
//
#include <hip/hip_runtime.h>
#include <hip/hip_bf16.h>
#include <math.h>

typedef __attribute__((ext_vector_type(16))) _Float16 v16h;
typedef __attribute__((ext_vector_type(8)))  _Float16 v8h;
typedef __attribute__((ext_vector_type(4)))  _Float16 v4h;
typedef __attribute__((ext_vector_type(8)))  float    v8f;

#define B_  2
#define S_  2048
#define D_  2048
#define HQ_ 16
#define HKV_ 4
#define HD_ 128
#define KVD_ 512
#define M_  (B_ * S_)   // 4096 rows for projections

static __device__ __forceinline__ v16h combine16(v8h lo, v8h hi) {
    v16h r;
#pragma unroll
    for (int i = 0; i < 8; ++i) { r[i] = lo[i]; r[i + 8] = hi[i]; }
    return r;
}

static __device__ __forceinline__ v8f wmma_f16(v16h a, v16h b, v8f c) {
    return __builtin_amdgcn_wmma_f32_16x16x32_f16(false, a, false, b, (short)0, c,
                                                  false, false);
}

static __device__ __forceinline__ float redmax16(float x) {
#pragma unroll
    for (int m = 1; m < 16; m <<= 1) x = fmaxf(x, __shfl_xor(x, m, 16));
    return x;
}
static __device__ __forceinline__ float redsum16(float x) {
#pragma unroll
    for (int m = 1; m < 16; m <<= 1) x += __shfl_xor(x, m, 16);
    return x;
}

// CDNA5 async copy: global memory -> LDS, 16 bytes per lane, tracked by ASYNCcnt.
// GVS addressing: mem = SGPR64 + VGPR32-offset; LDS dest offset from VGPR.
static __device__ __forceinline__ void async_b128(unsigned int lds_off,
                                                  unsigned int goff,
                                                  const void* sbase) {
    asm volatile("global_load_async_to_lds_b128 %0, %1, %2"
                 :: "v"(lds_off), "v"(goff), "s"(sbase)
                 : "memory");
}
static __device__ __forceinline__ void wait_async0() {
    asm volatile("s_wait_asynccnt 0x0" ::: "memory");
}
static __device__ __forceinline__ unsigned int lds_offset(const void* p) {
    return (unsigned int)(size_t)p;   // LDS aperture keeps offset in addr[31:0]
}

// ---------------------------------------------------------------------------
// One-time f32 -> f16 elementwise convert (hidden_states).
// ---------------------------------------------------------------------------
__global__ __launch_bounds__(256) void cvt_f16_kernel(const float* __restrict__ in,
                                                      _Float16* __restrict__ out) {
    const int i = blockIdx.x * 256 + threadIdx.x;
    const float4 f = ((const float4*)in)[i];
    v4h h;
    h[0] = (_Float16)f.x; h[1] = (_Float16)f.y;
    h[2] = (_Float16)f.z; h[3] = (_Float16)f.w;
    ((v4h*)out)[i] = h;
}

// ---------------------------------------------------------------------------
// One-time weight transpose + convert: Wt[n][k] = (f16) W[k][n].
// 32x32 tiles via LDS so both global read and write are coalesced.
// ---------------------------------------------------------------------------
__global__ __launch_bounds__(256) void wt_cvt_kernel(const float* __restrict__ W,
                                                     _Float16* __restrict__ Wt,
                                                     int N, int K) {
    __shared__ float tile[32][33];
    const int n0 = blockIdx.x * 32, k0 = blockIdx.y * 32;
    const int tn = threadIdx.x & 31, tk = threadIdx.x >> 5;
#pragma unroll
    for (int p = 0; p < 4; ++p)
        tile[tk + p * 8][tn] = W[(size_t)(k0 + tk + p * 8) * N + n0 + tn];
    __syncthreads();
#pragma unroll
    for (int p = 0; p < 4; ++p)
        Wt[(size_t)(n0 + tk + p * 8) * K + k0 + tn] =
            (_Float16)tile[tn][tk + p * 8];
}

// ---------------------------------------------------------------------------
// GEMM: C[M,N](f16) = Xh[M,K](f16) @ Wt[N,K]^T(f16) + bias(f32).
// Block tile 128x128, BK=32, double-buffered LDS filled by async-to-LDS copies.
// 8 waves: (wm 0..3)*32 rows x (wn 0..1)*64 cols, 8 WMMAs per wave per k-step.
// ---------------------------------------------------------------------------
__global__ __launch_bounds__(256) void gqa_gemm_kernel(
    const _Float16* __restrict__ X, const _Float16* __restrict__ Wt,
    const float* __restrict__ bias, _Float16* __restrict__ C, int N, int K) {

    // stride 40 halves (80 B) keeps 16-B alignment of fragment reads
    __shared__ _Float16 Xs[2][128 * 40];
    __shared__ _Float16 Ws[2][128 * 40];

    const int t = threadIdx.x;
    const int wave = t >> 5, lane = t & 31;
    const int half = lane >> 4, l = lane & 15;
    const int wm = wave & 3, wn = wave >> 2;
    const int m0 = blockIdx.x * 128, n0 = blockIdx.y * 128;

    const unsigned int xs_base[2] = { lds_offset(&Xs[0][0]), lds_offset(&Xs[1][0]) };
    const unsigned int ws_base[2] = { lds_offset(&Ws[0][0]), lds_offset(&Ws[1][0]) };

    // Issue one 128x32-half tile copy for X and Wt into LDS buffer `buf`.
    auto issue_tiles = [&](int buf, int k0) {
#pragma unroll
        for (int c = t; c < 512; c += 256) {          // X: 512 x 16B chunks
            const int m = c >> 2, part = c & 3;
            const unsigned int go =
                (unsigned int)(((m0 + m) * K + k0) * 2 + part * 16);
            async_b128(xs_base[buf] + (unsigned int)(m * 80 + part * 16), go, X);
        }
#pragma unroll
        for (int c = t; c < 512; c += 256) {          // Wt: 512 x 16B chunks
            const int n = c >> 2, part = c & 3;
            const unsigned int go =
                (unsigned int)(((n0 + n) * K + k0) * 2 + part * 16);
            async_b128(ws_base[buf] + (unsigned int)(n * 80 + part * 16), go, Wt);
        }
    };

    v8f acc[2][4];
#pragma unroll
    for (int mi = 0; mi < 2; ++mi)
#pragma unroll
        for (int ni = 0; ni < 4; ++ni) acc[mi][ni] = (v8f)0.0f;

    issue_tiles(0, 0);
    int cur = 0;
    for (int k0 = 0; k0 < K; k0 += 32) {
        wait_async0();          // own wave's copies into `cur` are done
        __syncthreads();        // all waves' copies into `cur` are done
        if (k0 + 32 < K) issue_tiles(cur ^ 1, k0 + 32);   // overlap with compute

        const _Float16* xs = &Xs[cur][0];
        const _Float16* ws = &Ws[cur][0];

        // A frags: lane<16 -> K0..7/K16..23 ; lane>=16 -> K8..15/K24..31
        v16h a[2];
#pragma unroll
        for (int mi = 0; mi < 2; ++mi) {
            const int row = wm * 32 + mi * 16 + l;
            v8h lo = *(const v8h*)&xs[row * 40 + half * 8];
            v8h hi = *(const v8h*)&xs[row * 40 + 16 + half * 8];
            a[mi] = combine16(lo, hi);
        }
        // B frags: lane<16 -> K0..7/K16..23 ; lane>=16 -> K8..15/K24..31
        v16h b[4];
#pragma unroll
        for (int ni = 0; ni < 4; ++ni) {
            const int col = wn * 64 + ni * 16 + l;
            v8h lo = *(const v8h*)&ws[col * 40 + half * 8];
            v8h hi = *(const v8h*)&ws[col * 40 + 16 + half * 8];
            b[ni] = combine16(lo, hi);
        }
#pragma unroll
        for (int mi = 0; mi < 2; ++mi)
#pragma unroll
            for (int ni = 0; ni < 4; ++ni)
                acc[mi][ni] = wmma_f16(a[mi], b[ni], acc[mi][ni]);

        __syncthreads();        // everyone done reading `cur` before it is refilled
        cur ^= 1;
    }

    // epilogue: C layout -> lane<16: M=r, lane>=16: M=r+8, N=l; store f16
#pragma unroll
    for (int mi = 0; mi < 2; ++mi) {
#pragma unroll
        for (int ni = 0; ni < 4; ++ni) {
            const int col = n0 + wn * 64 + ni * 16 + l;
            const float bb = bias[col];
#pragma unroll
            for (int r = 0; r < 8; ++r) {
                const int row = m0 + wm * 32 + mi * 16 + half * 8 + r;
                C[(size_t)row * N + col] = (_Float16)(acc[mi][ni][r] + bb);
            }
        }
    }
}

// ---------------------------------------------------------------------------
// Flash-style GQA attention on f16 Q/K/V. grid = (S/128, HQ, B), 8 waves.
// Each wave owns 16 query rows; key loop in tiles of 32 keys.
// K tile staged with async-to-LDS copies; V tile transposed into LDS.
// ---------------------------------------------------------------------------
__global__ __launch_bounds__(256) void gqa_attn_kernel(
    const _Float16* __restrict__ Q, const _Float16* __restrict__ Kbuf,
    const _Float16* __restrict__ Vbuf, float* __restrict__ out) {

    __shared__ _Float16 Ks[32 * 136];    // [key][hd], stride 136 (272B, 16B-aligned)
    __shared__ _Float16 Vts[128 * 40];   // [hd][key] transposed, stride 40
    __shared__ _Float16 Ps[8 * 16 * 40]; // per-wave P re-layout patch

    const int t = threadIdx.x;
    const int wave = t >> 5, lane = t & 31;
    const int half = lane >> 4, l = lane & 15;
    const int h = blockIdx.y, b = blockIdx.z;
    const int g = h & (HKV_ - 1);                 // kv head = h % HKV
    const int qbase = blockIdx.x * 128 + wave * 16;

    // ---- preload this wave's 16 queries as 4 A-frags, pre-scaled by 1/sqrt(HD)
    const float scale = 0.08838834764831845f;     // 1/sqrt(128)
    const _Float16* qp = Q + ((size_t)b * S_ + qbase + l) * D_ + h * HD_;
    v16h aq[4];
#pragma unroll
    for (int kc = 0; kc < 4; ++kc) {
        const int base = kc * 32 + half * 8;
        v16h a;
#pragma unroll
        for (int i = 0; i < 8; ++i)
            a[i] = (_Float16)((float)qp[base + i] * scale);
#pragma unroll
        for (int i = 0; i < 8; ++i)
            a[i + 8] = (_Float16)((float)qp[base + 16 + i] * scale);
        aq[kc] = a;
    }

    float mrow[8], lrow[8];
    v8f o[8];
#pragma unroll
    for (int r = 0; r < 8; ++r) { mrow[r] = -1e30f; lrow[r] = 0.0f; }
#pragma unroll
    for (int c = 0; c < 8; ++c) o[c] = (v8f)0.0f;

    const _Float16* kp = Kbuf + (size_t)b * S_ * KVD_ + g * HD_;
    const _Float16* vp = Vbuf + (size_t)b * S_ * KVD_ + g * HD_;
    _Float16* pw = &Ps[wave * 16 * 40];
    const unsigned int ks_base = lds_offset(&Ks[0]);

    for (int kb = 0; kb < S_; kb += 32) {
        // ---- K tile [32 keys][128 hd]: pure byte copy via async-to-LDS
#pragma unroll
        for (int c = t; c < 512; c += 256) {          // 512 x 16B chunks
            const int key = c >> 4, part = c & 15;
            const unsigned int go =
                (unsigned int)(((kb + key) * KVD_) * 2 + part * 16);
            async_b128(ks_base + (unsigned int)(key * 272 + part * 16), go, kp);
        }
        // ---- V tile transposed: Vts[hd][key] (f16, no conversion)
        {
            const int key = t >> 4, hd = (t & 15) * 8;
#pragma unroll
            for (int p = 0; p < 2; ++p) {
                const int kk = key + p * 16;
                const v8h v = *(const v8h*)(vp + (size_t)(kb + kk) * KVD_ + hd);
#pragma unroll
                for (int j = 0; j < 8; ++j) Vts[(hd + j) * 40 + kk] = v[j];
            }
        }
        wait_async0();
        __syncthreads();

        // ---- scores: 16 queries x 32 keys = two 16x16 WMMA tiles, K=128 (4x32)
        v8f s0 = (v8f)0.0f, s1 = (v8f)0.0f;
#pragma unroll
        for (int kc = 0; kc < 4; ++kc) {
            {   // keys 0..15
                const int krow = l;
                v8h lo = *(const v8h*)&Ks[krow * 136 + kc * 32 + half * 8];
                v8h hi = *(const v8h*)&Ks[krow * 136 + kc * 32 + 16 + half * 8];
                s0 = wmma_f16(aq[kc], combine16(lo, hi), s0);
            }
            {   // keys 16..31
                const int krow = 16 + l;
                v8h lo = *(const v8h*)&Ks[krow * 136 + kc * 32 + half * 8];
                v8h hi = *(const v8h*)&Ks[krow * 136 + kc * 32 + 16 + half * 8];
                s1 = wmma_f16(aq[kc], combine16(lo, hi), s1);
            }
        }

        // ---- online softmax (rows live across 16-lane groups)
        float alpha[8];
        v8f p0, p1;
#pragma unroll
        for (int r = 0; r < 8; ++r) {
            const float mx = redmax16(fmaxf(s0[r], s1[r]));
            const float mnew = fmaxf(mrow[r], mx);
            alpha[r] = __expf(mrow[r] - mnew);
            mrow[r] = mnew;
            const float e0 = __expf(s0[r] - mnew);
            const float e1 = __expf(s1[r] - mnew);
            p0[r] = e0; p1[r] = e1;
            lrow[r] = lrow[r] * alpha[r] + redsum16(e0 + e1);
        }
#pragma unroll
        for (int c = 0; c < 8; ++c)
#pragma unroll
            for (int r = 0; r < 8; ++r) o[c][r] *= alpha[r];

        // ---- re-layout P (C-layout -> A-layout) via per-wave LDS patch
#pragma unroll
        for (int r = 0; r < 8; ++r) {
            const int row = half * 8 + r;
            pw[row * 40 + l]      = (_Float16)p0[r];
            pw[row * 40 + 16 + l] = (_Float16)p1[r];
        }
        v8h plo = *(const v8h*)&pw[l * 40 + half * 8];
        v8h phi = *(const v8h*)&pw[l * 40 + 16 + half * 8];
        const v16h ap = combine16(plo, phi);

        // ---- O += P @ V  (8 column chunks of 16, K=32 keys)
#pragma unroll
        for (int c = 0; c < 8; ++c) {
            const int vrow = c * 16 + l;
            v8h lo = *(const v8h*)&Vts[vrow * 40 + half * 8];
            v8h hi = *(const v8h*)&Vts[vrow * 40 + 16 + half * 8];
            o[c] = wmma_f16(ap, combine16(lo, hi), o[c]);
        }
        __syncthreads();
    }

    // ---- epilogue: normalize and store ctx[b, q, h*128 + hd] (f32)
    float* op = out + (size_t)b * S_ * D_ + (size_t)h * HD_;
#pragma unroll
    for (int r = 0; r < 8; ++r) {
        const float inv = 1.0f / lrow[r];
        const int row = qbase + half * 8 + r;
#pragma unroll
        for (int c = 0; c < 8; ++c) {
            op[(size_t)row * D_ + c * 16 + l] = o[c][r] * inv;
        }
    }
}

extern "C" void kernel_launch(void* const* d_in, const int* in_sizes, int n_in,
                              void* d_out, int out_size, void* d_ws, size_t ws_size,
                              hipStream_t stream) {
    (void)in_sizes; (void)n_in; (void)out_size; (void)ws_size;
    const float* hs = (const float*)d_in[0];
    const float* Wq = (const float*)d_in[1];
    const float* bq = (const float*)d_in[2];
    const float* Wk = (const float*)d_in[3];
    const float* bk = (const float*)d_in[4];
    const float* Wv = (const float*)d_in[5];
    const float* bv = (const float*)d_in[6];

    _Float16* Xh  = (_Float16*)d_ws;                  // [4096][2048]
    _Float16* Wqt = Xh  + (size_t)M_ * D_;            // [2048][2048]  (Wt = W^T)
    _Float16* Wkt = Wqt + (size_t)D_ * D_;            // [512][2048]
    _Float16* Wvt = Wkt + (size_t)KVD_ * D_;          // [512][2048]
    _Float16* Qh  = Wvt + (size_t)KVD_ * D_;          // [4096][2048]
    _Float16* Kh  = Qh  + (size_t)M_ * D_;            // [4096][512]
    _Float16* Vh  = Kh  + (size_t)M_ * KVD_;          // [4096][512]

    dim3 blk(256);
    // one-time conversions / transposes
    cvt_f16_kernel<<<dim3((M_ * D_) / 1024), blk, 0, stream>>>(hs, Xh);
    wt_cvt_kernel<<<dim3(D_ / 32, D_ / 32), blk, 0, stream>>>(Wq, Wqt, D_, D_);
    wt_cvt_kernel<<<dim3(KVD_ / 32, D_ / 32), blk, 0, stream>>>(Wk, Wkt, KVD_, D_);
    wt_cvt_kernel<<<dim3(KVD_ / 32, D_ / 32), blk, 0, stream>>>(Wv, Wvt, KVD_, D_);
    // projections (f16 in, f16 out, f32 accumulate)
    gqa_gemm_kernel<<<dim3(M_ / 128, D_ / 128), blk, 0, stream>>>(Xh, Wqt, bq, Qh, D_, D_);
    gqa_gemm_kernel<<<dim3(M_ / 128, KVD_ / 128), blk, 0, stream>>>(Xh, Wkt, bk, Kh, KVD_, D_);
    gqa_gemm_kernel<<<dim3(M_ / 128, KVD_ / 128), blk, 0, stream>>>(Xh, Wvt, bv, Vh, KVD_, D_);
    // attention
    gqa_attn_kernel<<<dim3(S_ / 128, HQ_, B_), blk, 0, stream>>>(Qh, Kh, Vh, (float*)d_out);
}
